// FEMREncoderLayer_1159641170021
// MI455X (gfx1250) — matrix operand order repaired and checked by reading, not compiled, not run on hardware
//
#include <hip/hip_runtime.h>
#include <hip/hip_bf16.h>
#include <math.h>

#define H_DIM 768
#define I_DIM 3072
#define NHEAD 12
#define HS 64
#define NTOK 2048
#define WIN_N (3*H_DIM + 2*I_DIM)   /* 8448 */
#define CAT_N (H_DIM + I_DIM)       /* 3840 */

typedef _Float16 h16 __attribute__((ext_vector_type(16)));
typedef _Float16 h8  __attribute__((ext_vector_type(8)));
typedef float    f8  __attribute__((ext_vector_type(8)));
typedef unsigned u32x4 __attribute__((ext_vector_type(4)));
typedef int      i32x8 __attribute__((ext_vector_type(8)));
typedef int      i32x4 __attribute__((ext_vector_type(4)));

__device__ __forceinline__ unsigned lds_off(const void* p) {
  return (unsigned)(uintptr_t)p;   // low 32 bits of generic LDS address
}

// ---------------- async global -> LDS (ASYNCcnt path) ----------------
__device__ __forceinline__ void async_g2l_b128(unsigned lds_byte_off,
                                               const void* g) {
  asm volatile("global_load_async_to_lds_b128 %0, %1, off"
               :: "v"(lds_byte_off), "v"((unsigned long long)(uintptr_t)g)
               : "memory");
}
__device__ __forceinline__ void wait_async0() {
  asm volatile("s_wait_asynccnt 0" ::: "memory");
}

// ---------------- Tensor Data Mover 2D tile load (TENSORcnt path) ------------
// D# packing per CDNA5 ISA. Tile = tile_h lines of
// tile_w elements (2B each), line stride = stride_elems. LDS padding:
// pad_amount DWORD-code after every 2^pad_int DWORD-pairs -> padded LDS rows.
// This toolchain's builtin takes 6 args: (g0 v4u32, g1 v8i32, v4i32, v4i32,
// v8i32, i32 cpol) -- extra group zero-filled.
#define TDM_AVAILABLE __has_builtin(__builtin_amdgcn_tensor_load_to_lds)
__device__ __forceinline__ void tdm_load_2d(unsigned lds, const void* gaddr,
                                            unsigned tile_w, unsigned tile_h,
                                            unsigned long long stride_elems,
                                            unsigned pad_int, unsigned pad_amt) {
#if TDM_AVAILABLE
  const unsigned long long ga = (unsigned long long)(uintptr_t)gaddr;
  u32x4 g0; i32x8 g1; i32x4 g2 = {}; i32x4 g3 = {}; i32x8 g4 = {};
  g0[0] = 1u;                                     // count=1, user mode
  g0[1] = lds;                                    // lds_addr (bytes)
  g0[2] = (unsigned)ga;                           // global_addr[31:0]
  g0[3] = (unsigned)((ga >> 32) & 0x1FFFFFFu) | (2u << 30);  // addr hi | type=2
  g1[0] = (int)((1u << 16) |                      // data_size = 2 bytes
                (1u << 20) |                      // pad_enable
                (pad_int << 22) | (pad_amt << 25));
  g1[1] = (int)((tile_w & 0xFFFFu) << 16);        // tensor_dim0[15:0]
  g1[2] = (int)(((tile_w >> 16) & 0xFFFFu) | ((tile_h & 0xFFFFu) << 16));
  g1[3] = (int)(((tile_h >> 16) & 0xFFFFu) | ((tile_w & 0xFFFFu) << 16)); // tile_dim0
  g1[4] = (int)(tile_h & 0xFFFFu);                // tile_dim1 (tile_dim2=0)
  g1[5] = (int)(unsigned)(stride_elems & 0xFFFFFFFFu);       // dim0_stride lo
  g1[6] = (int)(unsigned)((stride_elems >> 32) & 0xFFFFu);   // dim0_stride hi
  g1[7] = 0;
  __builtin_amdgcn_tensor_load_to_lds(g0, g1, g2, g3, g4, 0);
#else
  (void)lds; (void)gaddr; (void)tile_w; (void)tile_h; (void)stride_elems;
  (void)pad_int; (void)pad_amt;
#endif
}
__device__ __forceinline__ void wait_tensor(int pending) {
#if TDM_AVAILABLE
  if (pending) __builtin_amdgcn_s_wait_tensorcnt(2);
  else         __builtin_amdgcn_s_wait_tensorcnt(0);
#else
  (void)pending;
#endif
}

// ---------------- 16-lane (half-wave) DPP ring reductions ----------------
#if __has_builtin(__builtin_amdgcn_update_dpp)
#define ROR_F32(v, n)                                                         \
  __builtin_bit_cast(float, __builtin_amdgcn_update_dpp(                      \
      0, __builtin_bit_cast(int, v), 0x120 + (n), 0xf, 0xf, true))
__device__ __forceinline__ float red16_add(float v) {
  v += ROR_F32(v, 1); v += ROR_F32(v, 2); v += ROR_F32(v, 4); v += ROR_F32(v, 8);
  return v;
}
__device__ __forceinline__ float red16_max(float v) {
  v = fmaxf(v, ROR_F32(v, 1)); v = fmaxf(v, ROR_F32(v, 2));
  v = fmaxf(v, ROR_F32(v, 4)); v = fmaxf(v, ROR_F32(v, 8));
  return v;
}
#else
__device__ __forceinline__ float red16_add(float v) {
  for (int m = 1; m <= 8; m <<= 1) v += __shfl_xor(v, m, 32);
  return v;
}
__device__ __forceinline__ float red16_max(float v) {
  for (int m = 1; m <= 8; m <<= 1) v = fmaxf(v, __shfl_xor(v, m, 32));
  return v;
}
#endif

// ---------------- WMMA fragment helpers (wave32, 16x16x32 f16) ----------------
__device__ __forceinline__ h16 frag_a(const _Float16* p, int ld) {
  const int lane = threadIdx.x & 31;
  const int m = lane & 15, hf = lane >> 4;
  h8 lo = *(const h8*)(p + m * ld + hf * 8);
  h8 hi = *(const h8*)(p + m * ld + hf * 8 + 16);
  return __builtin_shufflevector(lo, hi, 0, 1, 2, 3, 4, 5, 6, 7,
                                 8, 9, 10, 11, 12, 13, 14, 15);
}
__device__ __forceinline__ h16 frag_bT(const _Float16* p, int ld) {
  const int lane = threadIdx.x & 31;
  const int n = lane & 15, hf = lane >> 4;
  h8 lo = *(const h8*)(p + n * ld + hf * 8);
  h8 hi = *(const h8*)(p + n * ld + hf * 8 + 16);
  return __builtin_shufflevector(lo, hi, 0, 1, 2, 3, 4, 5, 6, 7,
                                 8, 9, 10, 11, 12, 13, 14, 15);
}
__device__ __forceinline__ f8 wmma16x16x32(h16 a, h16 b, f8 c) {
  return __builtin_amdgcn_wmma_f32_16x16x32_f16(false, a, false, b, (short)0, c,
                                                false, false);
}

// ---------------- fp32 -> f16 transposing convert (weights) ----------------
__global__ void k_cvt_t(const float* __restrict__ src, _Float16* __restrict__ dst,
                        int K, int N) {   // src: K x N  ->  dst: N x K
  const int i = blockIdx.x * 256 + threadIdx.x;
  if (i < K * N) {
    const int k = i / N, n = i - k * N;
    dst[(size_t)n * K + k] = (_Float16)src[i];
  }
}

// ---------------- RMSNorm + time override -> xn (f16) ----------------
__global__ __launch_bounds__(256)
void k_prep(const float* __restrict__ x, const float* __restrict__ td,
            const float* __restrict__ w, _Float16* __restrict__ xn) {
  const int n = blockIdx.x, tid = threadIdx.x;
  float vals[3];
  float ss = 0.f;
#pragma unroll
  for (int i = 0; i < 3; ++i) {
    vals[i] = x[(size_t)n * H_DIM + tid + i * 256];
    ss += vals[i] * vals[i];
  }
#pragma unroll
  for (int m = 16; m >= 1; m >>= 1) ss += __shfl_xor(ss, m, 32);
  __shared__ float wsum[8];
  if ((tid & 31) == 0) wsum[tid >> 5] = ss;
  __syncthreads();
  float tot = 0.f;
#pragma unroll
  for (int i = 0; i < 8; ++i) tot += wsum[i];
  const float rs = rsqrtf(tot * (1.0f / H_DIM) + 1e-6f);
  const float t0 = td[n * 2 + 0], t1 = td[n * 2 + 1];
#pragma unroll
  for (int i = 0; i < 3; ++i) {
    const int j = tid + i * 256;
    float o;
    if (j < H_DIM - 4) {
      o = vals[i] * rs * w[j];
    } else {
      const int c = j - (H_DIM - 4);
      o = (c == 0) ? t0 : (c == 1) ? t1 : (c == 2) ? t0 * t0 : t1 * t1;
    }
    xn[(size_t)n * H_DIM + j] = (_Float16)o;
  }
}

// ------- tiled f16 GEMM:  C(f32, M x NC) = A(f16 M x K) * BT(f16 NC x K) -----
// TDM staging, double-buffered; LDS row stride 40 f16 via TDM pad (16 DW + 4 DW)
#define SG_LD 40
template <int K, int NC>
__global__ __launch_bounds__(256)
void k_gemm_f16(const _Float16* __restrict__ A, const _Float16* __restrict__ BT,
                float* __restrict__ C) {
  __shared__ __align__(16) _Float16 sA[2][128 * SG_LD];
  __shared__ __align__(16) _Float16 sBT[2][128 * SG_LD];
  const int tid = threadIdx.x;
  const int lane = tid & 31, wv = tid >> 5;
  const int wm = wv >> 2, wn = wv & 3;          // 2x4 wave grid
  const int rowBase = blockIdx.y * 128;
  const int colBase = blockIdx.x * 128;
  const f8 zf = {};
  f8 acc[4][2];
#pragma unroll
  for (int i = 0; i < 4; ++i)
#pragma unroll
    for (int j = 0; j < 2; ++j) acc[i][j] = zf;

  const int nk = K / 32;
  if (wv == 0) {   // tile 0
    tdm_load_2d(lds_off(sA[0]), A + (size_t)rowBase * K, 32, 128, K, 3, 3);
    tdm_load_2d(lds_off(sBT[0]), BT + (size_t)colBase * K, 32, 128, K, 3, 3);
  }
  for (int i = 0; i < nk; ++i) {
    const int buf = i & 1;
    if (wv == 0) {
      if (i + 1 < nk) {
        const int k0n = (i + 1) * 32;
        tdm_load_2d(lds_off(sA[buf ^ 1]), A + (size_t)rowBase * K + k0n,
                    32, 128, K, 3, 3);
        tdm_load_2d(lds_off(sBT[buf ^ 1]), BT + (size_t)colBase * K + k0n,
                    32, 128, K, 3, 3);
        wait_tensor(1);     // <=2 outstanding -> tile i complete (in-order)
      } else {
        wait_tensor(0);
      }
    }
    __syncthreads();
    h16 bf[2];
#pragma unroll
    for (int nt = 0; nt < 2; ++nt)
      bf[nt] = frag_bT(sBT[buf] + (wn * 32 + nt * 16) * SG_LD, SG_LD);
#pragma unroll
    for (int mt = 0; mt < 4; ++mt) {
      h16 af = frag_a(sA[buf] + (wm * 64 + mt * 16) * SG_LD, SG_LD);
#pragma unroll
      for (int nt = 0; nt < 2; ++nt)
        acc[mt][nt] = wmma16x16x32(af, bf[nt], acc[mt][nt]);
    }
    __syncthreads();
  }
  const int hf = lane >> 4, nn = lane & 15;
  float* cp = C + (unsigned)(rowBase + wm * 64) * NC + colBase + wn * 32 + nn;
#pragma unroll
  for (int mt = 0; mt < 4; ++mt)
#pragma unroll
    for (int nt = 0; nt < 2; ++nt)
#pragma unroll
      for (int r = 0; r < 8; ++r)
        cp[(unsigned)(mt * 16 + r + 8 * hf) * NC + nt * 16] = acc[mt][nt][r];
}

// ------- RoPE(q,k), v pack (d-major!), silu(g1)*g2 -> concat ----------------
__global__ __launch_bounds__(256)
void k_rope_ff(const float* __restrict__ t, const float* __restrict__ psin,
               const float* __restrict__ pcos, _Float16* __restrict__ qh,
               _Float16* __restrict__ kh, _Float16* __restrict__ vh,
               _Float16* __restrict__ cat) {
  const int n = blockIdx.x, tid = threadIdx.x;
  const float* tr = t + (size_t)n * WIN_N;
  for (int j = tid; j < I_DIM; j += 256) {
    const float g1 = tr[j], g2 = tr[I_DIM + j];
    const float s = g1 / (1.0f + __expf(-g1));
    cat[(size_t)n * CAT_N + H_DIM + j] = (_Float16)(s * g2);
  }
  for (int idx = tid; idx < NHEAD * (HS / 2); idx += 256) {
    const int h = idx >> 5, d = (idx & 31) << 1;
    const float s0 = psin[n * HS + d], s1 = psin[n * HS + d + 1];
    const float c0 = pcos[n * HS + d], c1 = pcos[n * HS + d + 1];
    const int base = 2 * I_DIM + h * HS + d;
    const float q0 = tr[base], q1 = tr[base + 1];
    const float k0 = tr[base + H_DIM], k1 = tr[base + H_DIM + 1];
    const size_t o = ((size_t)h * NTOK + n) * HS + d;
    qh[o]     = (_Float16)(q0 * c0 - q1 * s0);
    qh[o + 1] = (_Float16)(q1 * c1 + q0 * s1);
    kh[o]     = (_Float16)(k0 * c0 - k1 * s0);
    kh[o + 1] = (_Float16)(k1 * c1 + k0 * s1);
  }
  for (int j = tid; j < H_DIM; j += 256) {    // v stored [NH][HS][NTOK]
    const int h = j >> 6, d = j & 63;
    vh[((size_t)h * HS + d) * NTOK + n] =
        (_Float16)tr[2 * I_DIM + 2 * H_DIM + j];
  }
}

// ---------------- flash attention per (head, 64-query tile) ----------------
// K/V tiles via TDM (double-buffered), Q via async b128, softmax via DPP.
#define AT_LD 72
__global__ __launch_bounds__(128)
void k_attn(const _Float16* __restrict__ qh, const _Float16* __restrict__ kh,
            const _Float16* __restrict__ vh, const float* __restrict__ bias,
            _Float16* __restrict__ cat) {
  __shared__ __align__(16) _Float16 sQ[64 * AT_LD];
  __shared__ __align__(16) _Float16 sK[2][64 * AT_LD];
  __shared__ __align__(16) _Float16 sVT[2][64 * AT_LD];   // [d][key]
  __shared__ __align__(16) _Float16 sP[64 * AT_LD];
  const int tid = threadIdx.x, lane = tid & 31, wv = tid >> 5;
  const int hf = lane >> 4, nn = lane & 15;
  const int h = blockIdx.y;
  const int qBase = blockIdx.x * 64;
  const _Float16* qp = qh + ((size_t)h * NTOK + qBase) * HS;
  const _Float16* kp = kh + (size_t)h * NTOK * HS;
  const _Float16* vtp = vh + (size_t)h * HS * NTOK;
  const float* bp = bias + ((size_t)h * NTOK + qBase) * NTOK;

  const unsigned sQo = lds_off(sQ);
#pragma unroll
  for (int t = 0; t < 4; ++t) {              // Q tile 64 x 64 (ASYNCcnt path)
    const int c = t * 128 + tid;
    const int r = c >> 3, cc = (c & 7) << 3;
    async_g2l_b128(sQo + (unsigned)(r * AT_LD + cc) * 2, qp + r * HS + cc);
  }
  wait_async0();

  if (wv == 0) {                             // K/V tile 0 (TENSORcnt path)
    tdm_load_2d(lds_off(sK[0]), kp, 64, 64, HS, 4, 3);
    tdm_load_2d(lds_off(sVT[0]), vtp, 64, 64, NTOK, 4, 3);
  }

  const f8 zf = {};
  f8 o[4];
  float mrow[8], lrow[8];
#pragma unroll
  for (int i = 0; i < 4; ++i) o[i] = zf;
#pragma unroll
  for (int r = 0; r < 8; ++r) { mrow[r] = -1.0e30f; lrow[r] = 0.f; }

  for (int kb = 0; kb < NTOK; kb += 64) {
    const int buf = (kb >> 6) & 1;
    if (wv == 0) {
      if (kb + 64 < NTOK) {
        tdm_load_2d(lds_off(sK[buf ^ 1]), kp + (size_t)(kb + 64) * HS,
                    64, 64, HS, 4, 3);
        tdm_load_2d(lds_off(sVT[buf ^ 1]), vtp + kb + 64,
                    64, 64, NTOK, 4, 3);
        wait_tensor(1);
      } else {
        wait_tensor(0);
      }
    }
    __syncthreads();
    // S = Q * K^T
    f8 s[4];
#pragma unroll
    for (int i = 0; i < 4; ++i) s[i] = zf;
#pragma unroll
    for (int kk = 0; kk < 64; kk += 32) {
      h16 aq = frag_a(sQ + (wv * 16) * AT_LD + kk, AT_LD);
#pragma unroll
      for (int nt = 0; nt < 4; ++nt) {
        h16 bk = frag_bT(sK[buf] + (nt * 16) * AT_LD + kk, AT_LD);
        s[nt] = wmma16x16x32(aq, bk, s[nt]);
      }
    }
    // bias stream from HBM into the accumulator fragments
#pragma unroll
    for (int nt = 0; nt < 4; ++nt)
#pragma unroll
      for (int r = 0; r < 8; ++r) {
        const int qr = wv * 16 + r + 8 * hf;
        const int kc = kb + nt * 16 + nn;
        s[nt][r] = s[nt][r] * 0.125f + bp[(unsigned)qr * NTOK + kc];
      }
    if (kb + 64 < NTOK)
      __builtin_prefetch(bp + (size_t)(wv * 16 + (tid & 15)) * NTOK + kb + 64,
                         0, 1);
    // online softmax; rows r / r+8 live in disjoint 16-lane halves (DPP rows)
#pragma unroll
    for (int r = 0; r < 8; ++r) {
      const float rmax =
          red16_max(fmaxf(fmaxf(s[0][r], s[1][r]), fmaxf(s[2][r], s[3][r])));
      const float mn = fmaxf(mrow[r], rmax);
      const float alpha = __expf(mrow[r] - mn);
      mrow[r] = mn;
      float rsum = 0.f;
#pragma unroll
      for (int nt = 0; nt < 4; ++nt) {
        const float p = __expf(s[nt][r] - mn);
        rsum += p;
        sP[(wv * 16 + r + 8 * hf) * AT_LD + nt * 16 + nn] = (_Float16)p;
      }
      lrow[r] = lrow[r] * alpha + red16_add(rsum);
#pragma unroll
      for (int nt = 0; nt < 4; ++nt) o[nt][r] *= alpha;
    }
    // O += P * V   (P staged in LDS; same-wave LDS RAW is in-order)
#pragma unroll
    for (int kk = 0; kk < 64; kk += 32) {
      h16 ap = frag_a(sP + (wv * 16) * AT_LD + kk, AT_LD);
#pragma unroll
      for (int nt = 0; nt < 4; ++nt) {
        h16 bv = frag_bT(sVT[buf] + (nt * 16) * AT_LD + kk, AT_LD);
        o[nt] = wmma16x16x32(ap, bv, o[nt]);
      }
    }
    __syncthreads();
  }
#pragma unroll
  for (int nt = 0; nt < 4; ++nt)
#pragma unroll
    for (int r = 0; r < 8; ++r) {
      const int n_tok = qBase + wv * 16 + r + 8 * hf;
      const int d = nt * 16 + nn;
      cat[(size_t)n_tok * CAT_N + h * HS + d] =
          (_Float16)(o[nt][r] / lrow[r]);
    }
}

// ---------------- host side ----------------
extern "C" void kernel_launch(void* const* d_in, const int* in_sizes, int n_in,
                              void* d_out, int out_size, void* d_ws,
                              size_t ws_size, hipStream_t stream) {
  (void)in_sizes; (void)n_in; (void)out_size; (void)ws_size;
  const float* x  = (const float*)d_in[0];
  const float* td = (const float*)d_in[1];
  const float* ps = (const float*)d_in[2];
  const float* pc = (const float*)d_in[3];
  const float* ab = (const float*)d_in[4];
  const float* nw = (const float*)d_in[5];
  const float* wi = (const float*)d_in[6];
  const float* wo = (const float*)d_in[7];
  float* out = (float*)d_out;

  char* ws = (char*)d_ws;
  auto alloc = [&](size_t bytes) {
    char* p = ws;
    ws += (bytes + 255) & ~(size_t)255;
    return p;
  };
  _Float16* xn   = (_Float16*)alloc((size_t)NTOK * H_DIM * 2);
  _Float16* wi_t = (_Float16*)alloc((size_t)H_DIM * WIN_N * 2);  // [WIN_N][H]
  _Float16* wo_t = (_Float16*)alloc((size_t)CAT_N * H_DIM * 2);  // [H][CAT_N]
  float*    tbuf = (float*)   alloc((size_t)NTOK * WIN_N * 4);
  _Float16* qh   = (_Float16*)alloc((size_t)NHEAD * NTOK * HS * 2);
  _Float16* kh   = (_Float16*)alloc((size_t)NHEAD * NTOK * HS * 2);
  _Float16* vh   = (_Float16*)alloc((size_t)NHEAD * NTOK * HS * 2);
  _Float16* cat  = (_Float16*)alloc((size_t)NTOK * CAT_N * 2);

  k_cvt_t<<<(H_DIM * WIN_N + 255) / 256, 256, 0, stream>>>(wi, wi_t, H_DIM,
                                                           WIN_N);
  k_cvt_t<<<(CAT_N * H_DIM + 255) / 256, 256, 0, stream>>>(wo, wo_t, CAT_N,
                                                           H_DIM);
  k_prep<<<NTOK, 256, 0, stream>>>(x, td, nw, xn);
  k_gemm_f16<H_DIM, WIN_N>
      <<<dim3(WIN_N / 128, NTOK / 128), 256, 0, stream>>>(xn, wi_t, tbuf);
  k_rope_ff<<<NTOK, 256, 0, stream>>>(tbuf, ps, pc, qh, kh, vh, cat);
  k_attn<<<dim3(NTOK / 64, NHEAD), 128, 0, stream>>>(qh, kh, vh, ab, cat);
  k_gemm_f16<CAT_N, H_DIM>
      <<<dim3(H_DIM / 128, NTOK / 128), 256, 0, stream>>>(cat, wo_t, out);
}
